// CustomLSTM_11905649345052
// MI455X (gfx1250) — matrix operand order repaired and checked by reading, not compiled, not run on hardware
//
#include <hip/hip_runtime.h>

typedef __attribute__((ext_vector_type(16))) __bf16 v16bf;
typedef __attribute__((ext_vector_type(8)))  __bf16 v8bf;
typedef __attribute__((ext_vector_type(8)))  float  v8f;

#define DEVINL __device__ __forceinline__
#define USE_ASYNC 1   // CDNA5 global_load_async_to_lds_* staging (set 0 to fall back)

// ---------- helpers ----------
DEVINL unsigned short f2bf(float f) {
    unsigned u = __float_as_uint(f);
    unsigned v = u + 0x7FFFu + ((u >> 16) & 1u);   // round-to-nearest-even
    return (unsigned short)(v >> 16);
}

DEVINL v16bf pack16(v8bf lo, v8bf hi) {
    v16bf r;
#pragma unroll
    for (int i = 0; i < 8; ++i) { r[i] = lo[i]; r[i + 8] = hi[i]; }
    return r;
}

#if USE_ASYNC
// CDNA5 async copy: 16B global -> LDS per lane, tracked by ASYNCcnt (no VGPR data path)
DEVINL void async_g2l_b128(const void* gsrc, void* ldst) {
    unsigned loff = (unsigned)(unsigned long long)ldst;       // low 32b of flat shared ptr = LDS offset
    unsigned long long ga = (unsigned long long)gsrc;
    asm volatile("global_load_async_to_lds_b128 %0, %1, off" :: "v"(loff), "v"(ga) : "memory");
}
DEVINL void wait_async0() { asm volatile("s_wait_asynccnt 0" ::: "memory"); }
#else
DEVINL void async_g2l_b128(const void* gsrc, void* ldst) {
    *(uint4*)ldst = *(const uint4*)gsrc;
}
DEVINL void wait_async0() {}
#endif

// fp32 -> bf16 (2 elements per thread, packed 32-bit store)
__global__ __launch_bounds__(256) void cvt_bf16_kernel(const float* __restrict__ in,
                                                       unsigned short* __restrict__ out,
                                                       int n2) {
    int i = blockIdx.x * 256 + threadIdx.x;
    if (i < n2) {
        float2 v = ((const float2*)in)[i];
        unsigned p = (unsigned)f2bf(v.x) | ((unsigned)f2bf(v.y) << 16);
        ((unsigned*)out)[i] = p;
    }
}

// ---------- big input-projection GEMM ----------
// C[M,N] f32 = A[M,K] bf16  x  W[N,K] bf16 (as B^T)  + (b0[N] + b1[N])
// Block: 256 thr = 8 waves, C tile 128x128, K-step 32.
// Explicit double-buffered async global->LDS staging: DMA of tile k+1
// overlaps the 8 WMMAs of tile k; one barrier per k-step.
__global__ __launch_bounds__(256) void gemm_bf16_kernel(
    const unsigned short* __restrict__ A,
    const unsigned short* __restrict__ W,
    const float* __restrict__ b0, const float* __restrict__ b1,
    float* __restrict__ C, int M, int N, int K) {
    __shared__ unsigned short As[2][128 * 40];   // padded stride 40 halves
    __shared__ unsigned short Bs[2][128 * 40];

    const int nb   = N >> 7;
    const int bm   = blockIdx.x / nb, bn = blockIdx.x % nb;
    const int m0   = bm << 7, n0 = bn << 7;
    const int tid  = threadIdx.x;
    const int w    = tid >> 5, lane = tid & 31;
    const int lm   = lane & 15;
    const int hi8  = (lane >> 4) << 3;        // 0 or 8
    const int ksB  = (lane < 16) ? 0 : 8;
    const int wm   = (w >> 1) << 5;           // 0,32,64,96
    const int wn   = (w & 1)  << 6;           // 0,64

    // per-thread staging coordinates (2 x 16B chunks per matrix per k-step)
    const int idx0 = tid * 2, idx1 = tid * 2 + 1;
    const int r0 = idx0 >> 2, ko0 = (idx0 & 3) << 3;
    const int r1 = idx1 >> 2, ko1 = (idx1 & 3) << 3;

    v8f acc[2][4];
#pragma unroll
    for (int i = 0; i < 2; ++i)
#pragma unroll
        for (int j = 0; j < 4; ++j)
#pragma unroll
            for (int v = 0; v < 8; ++v) acc[i][j][v] = 0.f;

    // prologue: stage tile 0
    async_g2l_b128(&A[(size_t)(m0 + r0) * K + ko0], &As[0][r0 * 40 + ko0]);
    async_g2l_b128(&W[(size_t)(n0 + r0) * K + ko0], &Bs[0][r0 * 40 + ko0]);
    async_g2l_b128(&A[(size_t)(m0 + r1) * K + ko1], &As[0][r1 * 40 + ko1]);
    async_g2l_b128(&W[(size_t)(n0 + r1) * K + ko1], &Bs[0][r1 * 40 + ko1]);
    wait_async0();
    __syncthreads();

    int cur = 0;
    for (int k0 = 0; k0 < K; k0 += 32) {
        const int nxt = cur ^ 1;
        if (k0 + 32 < K) {  // issue DMA for next tile; lands during this tile's WMMAs
            const int kn = k0 + 32;
            async_g2l_b128(&A[(size_t)(m0 + r0) * K + kn + ko0], &As[nxt][r0 * 40 + ko0]);
            async_g2l_b128(&W[(size_t)(n0 + r0) * K + kn + ko0], &Bs[nxt][r0 * 40 + ko0]);
            async_g2l_b128(&A[(size_t)(m0 + r1) * K + kn + ko1], &As[nxt][r1 * 40 + ko1]);
            async_g2l_b128(&W[(size_t)(n0 + r1) * K + kn + ko1], &Bs[nxt][r1 * 40 + ko1]);
        }
#pragma unroll
        for (int mt = 0; mt < 2; ++mt) {
            int mrow = wm + (mt << 4) + lm;
            v8bf a0 = *(const v8bf*)&As[cur][mrow * 40 + hi8];
            v8bf a1 = *(const v8bf*)&As[cur][mrow * 40 + hi8 + 16];
            v16bf af = pack16(a0, a1);
#pragma unroll
            for (int nt = 0; nt < 4; ++nt) {
                int ncol = wn + (nt << 4) + lm;
                v8bf bl = *(const v8bf*)&Bs[cur][ncol * 40 + ksB];
                v8bf bh = *(const v8bf*)&Bs[cur][ncol * 40 + ksB + 16];
                acc[mt][nt] = __builtin_amdgcn_wmma_f32_16x16x32_bf16(
                    false, af, false, pack16(bl, bh), (short)0, acc[mt][nt], false, false);
            }
        }
        wait_async0();       // next tile landed
        __syncthreads();     // everyone done reading cur
        cur = nxt;
    }
#pragma unroll
    for (int mt = 0; mt < 2; ++mt)
#pragma unroll
        for (int nt = 0; nt < 4; ++nt) {
            int ncol = n0 + wn + (nt << 4) + lm;
            float bias = b0[ncol] + b1[ncol];
#pragma unroll
            for (int v = 0; v < 8; ++v) {
                int mg = m0 + wm + (mt << 4) + v + hi8;
                C[(size_t)mg * N + ncol] = acc[mt][nt][v] + bias;
            }
        }
}

// ---------- software grid barrier (per-wave acquire => per-CU L0 invalidation) ----------
DEVINL void grid_barrier(unsigned* bar, unsigned nwg) {
    __syncthreads();
    const bool lead = ((threadIdx.x & 31) == 0);
    unsigned my_gen = 0;
    if (lead)
        my_gen = __hip_atomic_load(&bar[1], __ATOMIC_ACQUIRE, __HIP_MEMORY_SCOPE_AGENT);
    __syncthreads();   // all leaders captured gen before this WG is counted
    if (threadIdx.x == 0) {
        unsigned prev = __hip_atomic_fetch_add(&bar[0], 1u, __ATOMIC_ACQ_REL, __HIP_MEMORY_SCOPE_AGENT);
        if (prev == nwg - 1) {
            __hip_atomic_store(&bar[0], 0u, __ATOMIC_RELAXED, __HIP_MEMORY_SCOPE_AGENT);
            __hip_atomic_fetch_add(&bar[1], 1u, __ATOMIC_RELEASE, __HIP_MEMORY_SCOPE_AGENT);
        }
    }
    if (lead) {
        while (__hip_atomic_load(&bar[1], __ATOMIC_ACQUIRE, __HIP_MEMORY_SCOPE_AGENT) == my_gen)
            __builtin_amdgcn_s_sleep(1);
    }
    __syncthreads();
}

// ---------- persistent recurrent LSTM scan ----------
// grid = 32 WGs: dir = blk/16, g = blk%16; WG owns hidden units [g*32, g*32+32)
// => 128 gate columns (gate-major local layout: col = gate*32 + unit).
// W_hh slice LDS-resident; h staged to LDS each step via async copy from ping-pong bufs.
#define WS_STRIDE 520
#define HS_STRIDE 520
#define SMEM_RECUR (128 * WS_STRIDE * 2 + 64 * HS_STRIDE * 2)   // 199680 B

__global__ __launch_bounds__(256) void lstm_recur_kernel(
    const unsigned short* __restrict__ whh,  // [2][2048][512] bf16 (dir-major)
    const float* __restrict__ xpF,           // [T*64][2048] f32
    const float* __restrict__ xpB,
    unsigned short* __restrict__ hbuf,       // [2 dir][2 pp][64*512] bf16
    unsigned short* __restrict__ out_bf,     // [T*64][1024] bf16 or null
    float* __restrict__ out_f32,             // [T*64][1024] f32 or null
    float* __restrict__ hn, float* __restrict__ cn,  // [2][64][512] (this layer)
    unsigned* __restrict__ bar, int T) {
    extern __shared__ char smem[];
    unsigned short* Ws = (unsigned short*)smem;                          // 128 x 520
    unsigned short* Hs = (unsigned short*)(smem + 128 * WS_STRIDE * 2);  // 64 x 520
    float*          Lg = (float*)(smem + 128 * WS_STRIDE * 2);           // reuse: 64 x 132

    const int dir = blockIdx.x >> 4;
    const int g   = blockIdx.x & 15;
    const int tid = threadIdx.x;
    const int w   = tid >> 5, lane = tid & 31;
    const int lm  = lane & 15;
    const int hi8 = (lane >> 4) << 3;
    const int ksB = (lane < 16) ? 0 : 8;
    const int gate = w >> 1;                       // uniform per wave
    const int unit = ((w & 1) << 4) + lm;          // 0..31
    const int jg   = gate * 512 + g * 32 + unit;   // global gate column
    const int cloc = (w << 4) + lm;                // local col 0..127
    const unsigned nwg = gridDim.x;
    const float* xp = dir ? xpB : xpF;

    // stage W_hh slice into LDS once (async; waited with first h stage)
    const unsigned short* whhd = whh + (size_t)dir * 2048 * 512;
    for (int idx = tid; idx < 128 * 64; idx += 256) {
        int c = idx >> 6, ko = (idx & 63) << 3;
        int j = (c >> 5) * 512 + g * 32 + (c & 31);
        async_g2l_b128(&whhd[(size_t)j * 512 + ko], &Ws[c * WS_STRIDE + ko]);
    }

    float cReg[8];
#pragma unroll
    for (int e = 0; e < 8; ++e) cReg[e] = 0.f;

    for (int s = 0; s < T; ++s) {
        const int pp = s & 1, ppr = pp ^ 1;
        // stage h(s-1) -> LDS (async, bypasses VGPRs)
        const unsigned short* hsrc = hbuf + ((size_t)dir * 2 + ppr) * (64 * 512);
        for (int idx = tid; idx < 4096; idx += 256) {
            int r = idx >> 6, ko = (idx & 63) << 3;
            async_g2l_b128(&hsrc[r * 512 + ko], &Hs[r * HS_STRIDE + ko]);
        }
        const int t = dir ? (T - 1 - s) : s;
        const float* xpt = xp + (size_t)t * 64 * 2048;

        // prefetch next step's xp slice while async staging is in flight
        if (s + 1 < T) {
            const float* xpn = xp + (size_t)(dir ? (T - 2 - s) : (s + 1)) * 64 * 2048;
#pragma unroll
            for (int mt = 0; mt < 4; ++mt)
                __builtin_prefetch(&xpn[(size_t)((mt << 4) + hi8) * 2048 + jg], 0, 1);
        }

        // gate-bias accumulators: global fp32 loads in flight during async staging
        v8f acc[4];
#pragma unroll
        for (int mt = 0; mt < 4; ++mt)
#pragma unroll
            for (int v = 0; v < 8; ++v)
                acc[mt][v] = xpt[(size_t)((mt << 4) + v + hi8) * 2048 + jg];

        wait_async0();
        __syncthreads();

#pragma unroll 4
        for (int kt = 0; kt < 16; ++kt) {
            v8bf bl = *(const v8bf*)&Ws[cloc * WS_STRIDE + (kt << 5) + ksB];
            v8bf bh = *(const v8bf*)&Ws[cloc * WS_STRIDE + (kt << 5) + ksB + 16];
            v16bf bf_ = pack16(bl, bh);
#pragma unroll
            for (int mt = 0; mt < 4; ++mt) {
                int m = (mt << 4) + lm;
                v8bf a0 = *(const v8bf*)&Hs[m * HS_STRIDE + (kt << 5) + hi8];
                v8bf a1 = *(const v8bf*)&Hs[m * HS_STRIDE + (kt << 5) + hi8 + 16];
                acc[mt] = __builtin_amdgcn_wmma_f32_16x16x32_bf16(
                    false, pack16(a0, a1), false, bf_, (short)0, acc[mt], false, false);
            }
        }
        __syncthreads();   // done reading Hs; reuse as Lg

        const bool is_tanh = (gate == 2);
#pragma unroll
        for (int mt = 0; mt < 4; ++mt)
#pragma unroll
            for (int v = 0; v < 8; ++v) {
                float x = acc[mt][v];
                float a = is_tanh ? tanhf(x) : (1.f / (1.f + __expf(-x)));
                Lg[((mt << 4) + v + hi8) * 132 + cloc] = a;
            }
        __syncthreads();

        // c/h update: 2048 (row,unit) elems, 8 per thread; c lives in registers
        unsigned short* hdst = hbuf + ((size_t)dir * 2 + pp) * (64 * 512);
#pragma unroll
        for (int e = 0; e < 8; ++e) {
            int linear = e * 256 + tid;
            int row = linear >> 5, u = linear & 31;
            float iv = Lg[row * 132 + u];
            float fv = Lg[row * 132 + 32 + u];
            float gv = Lg[row * 132 + 64 + u];
            float ov = Lg[row * 132 + 96 + u];
            float cc = fv * cReg[e] + iv * gv;
            cReg[e] = cc;
            float hh = ov * tanhf(cc);
            int hcol = g * 32 + u;
            hdst[row * 512 + hcol] = f2bf(hh);
            size_t orow = (size_t)t * 64 + row;
            if (out_bf)  out_bf[orow * 1024 + dir * 512 + hcol] = f2bf(hh);
            if (out_f32) out_f32[orow * 1024 + dir * 512 + hcol] = hh;
            if (s == T - 1) {
                hn[((size_t)dir * 64 + row) * 512 + hcol] = hh;
                cn[((size_t)dir * 64 + row) * 512 + hcol] = cc;
            }
        }
        __threadfence();
        grid_barrier(bar, nwg);
    }
}

// ---------- host ----------
extern "C" void kernel_launch(void* const* d_in, const int* in_sizes, int n_in,
                              void* d_out, int out_size, void* d_ws, size_t ws_size,
                              hipStream_t stream) {
    (void)in_sizes; (void)n_in; (void)out_size; (void)ws_size;
    constexpr int Tn = 1024, Bn = 64, In = 512, Hn = 512, FH = 2048;
    constexpr size_t M = (size_t)Tn * Bn;              // 65536

    const float* x     = (const float*)d_in[0];
    const float* w_ih0 = (const float*)d_in[1];
    const float* w_hh0 = (const float*)d_in[2];
    const float* b_ih0 = (const float*)d_in[3];
    const float* b_hh0 = (const float*)d_in[4];
    const float* w_ih1 = (const float*)d_in[5];
    const float* w_hh1 = (const float*)d_in[6];
    const float* b_ih1 = (const float*)d_in[7];
    const float* b_hh1 = (const float*)d_in[8];

    // workspace layout
    char* ws = (char*)d_ws;
    size_t off = 0;
    unsigned*       bar    = (unsigned*)(ws + off);       off += 256;
    unsigned short* hbuf   = (unsigned short*)(ws + off); off += (size_t)2 * 2 * Bn * Hn * 2;   // 262144
    size_t ctl_bytes = off;                               // memset region: barrier + hbuf
    unsigned short* xbf    = (unsigned short*)(ws + off); off += M * In * 2;
    unsigned short* wih0bf = (unsigned short*)(ws + off); off += (size_t)2 * FH * In * 2;
    unsigned short* whh0bf = (unsigned short*)(ws + off); off += (size_t)2 * FH * Hn * 2;
    unsigned short* wih1bf = (unsigned short*)(ws + off); off += (size_t)2 * FH * (2 * Hn) * 2;
    unsigned short* whh1bf = (unsigned short*)(ws + off); off += (size_t)2 * FH * Hn * 2;
    unsigned short* out0bf = (unsigned short*)(ws + off); off += M * (2 * Hn) * 2;
    float*          xpf    = (float*)(ws + off);          off += M * FH * 4;
    float*          xpb    = (float*)(ws + off);          off += M * FH * 4;

    float* out1 = (float*)d_out;
    float* hn0  = out1 + M * (2 * Hn);                    // 4 x 64 x 512
    float* cn0  = hn0 + (size_t)4 * Bn * Hn;

    hipMemsetAsync(d_ws, 0, ctl_bytes, stream);           // barrier + h ping-pong = 0

    // fp32 -> bf16 conversions
    auto cvt = [&](const float* src, unsigned short* dst, size_t n) {
        int n2 = (int)(n / 2);
        cvt_bf16_kernel<<<(n2 + 255) / 256, 256, 0, stream>>>(src, dst, n2);
    };
    cvt(x,     xbf,    M * In);
    cvt(w_ih0, wih0bf, (size_t)2 * FH * In);
    cvt(w_hh0, whh0bf, (size_t)2 * FH * Hn);
    cvt(w_ih1, wih1bf, (size_t)2 * FH * 2 * Hn);
    cvt(w_hh1, whh1bf, (size_t)2 * FH * Hn);

    const dim3 gblk(256);
    // layer-0 input projections (K = 512)
    {
        dim3 grid((unsigned)((M / 128) * (FH / 128)));
        gemm_bf16_kernel<<<grid, gblk, 0, stream>>>(xbf, wih0bf,                  b_ih0,      b_hh0,      xpf, (int)M, FH, In);
        gemm_bf16_kernel<<<grid, gblk, 0, stream>>>(xbf, wih0bf + (size_t)FH * In, b_ih0 + FH, b_hh0 + FH, xpb, (int)M, FH, In);
    }
    // layer-0 recurrence (writes out0 as bf16 + h_n/c_n rows 0,1)
    lstm_recur_kernel<<<32, 256, SMEM_RECUR, stream>>>(whh0bf, xpf, xpb, hbuf,
                                                       out0bf, nullptr, hn0, cn0, bar, Tn);
    // reset barrier + h ping-pong for layer 1
    hipMemsetAsync(d_ws, 0, ctl_bytes, stream);
    // layer-1 input projections (K = 1024)
    {
        dim3 grid((unsigned)((M / 128) * (FH / 128)));
        gemm_bf16_kernel<<<grid, gblk, 0, stream>>>(out0bf, wih1bf,                        b_ih1,      b_hh1,      xpf, (int)M, FH, 2 * Hn);
        gemm_bf16_kernel<<<grid, gblk, 0, stream>>>(out0bf, wih1bf + (size_t)FH * 2 * Hn,  b_ih1 + FH, b_hh1 + FH, xpb, (int)M, FH, 2 * Hn);
    }
    // layer-1 recurrence (writes final fp32 output + h_n/c_n rows 2,3)
    lstm_recur_kernel<<<32, 256, SMEM_RECUR, stream>>>(whh1bf, xpf, xpb, hbuf,
                                                       nullptr, out1,
                                                       hn0 + (size_t)2 * Bn * Hn,
                                                       cn0 + (size_t)2 * Bn * Hn, bar, Tn);
}
